// precalcSDE_35218731827223
// MI455X (gfx1250) — compile-verified, matrix operand order checked by reading, not run on hardware
//
#include <hip/hip_runtime.h>
#include <hip/hip_bf16.h>
#include <math.h>

typedef __attribute__((ext_vector_type(16))) _Float16 v16h;
typedef __attribute__((ext_vector_type(8)))  _Float16 v8h;   // 16 B
typedef __attribute__((ext_vector_type(8)))  float    v8f;
typedef __attribute__((ext_vector_type(4)))  int      v4i;   // 16 B

#define V_WORDS   32000
#define NSPEC     4
#define DIM       128
#define LATENT    10000
#define LATENT_P  10016          // padded to multiple of 32
#define NCHUNK    (LATENT_P / 32)
#define LOG2E     1.44269504088896340736f

// Padded LDS strides (halfs): 16-B aligned rows, conflict-spread lanes
#define BS_STRIDE   136          // 272 B = 17 * 16
#define BST_STRIDE  40           // 80 B  =  5 * 16
#define P_STRIDE    40

// -------- async global->LDS path (gfx1250), with safe fallback ------------
#if __has_builtin(__builtin_amdgcn_global_load_async_to_lds_b128) && \
    __has_builtin(__builtin_amdgcn_s_wait_asynccnt)
#define USE_ASYNC 1
typedef __attribute__((address_space(1))) v4i* gv4i_p;  // global v4i*
typedef __attribute__((address_space(3))) v4i* lv4i_p;  // LDS    v4i*
#else
#define USE_ASYNC 0
#endif

__device__ __forceinline__ void stage16(const _Float16* g, _Float16* l) {
#if USE_ASYNC
    __builtin_amdgcn_global_load_async_to_lds_b128((gv4i_p)g, (lv4i_p)l, 0, 0);
#else
    *(v8h*)l = *(const v8h*)g;
#endif
}

__device__ __forceinline__ void wait_async_lds() {
#if USE_ASYNC
    __builtin_amdgcn_s_wait_asynccnt(0);
#endif
}

// ---------------------------------------------------------------------------
// Kernel 1: EmbeddingBag(sum) over sorted segment ids + tanh.
// ---------------------------------------------------------------------------
__device__ __forceinline__ int lower_bound_i32(const int* a, int n, int key) {
    int lo = 0, hi = n;
    while (lo < hi) {
        int mid = (lo + hi) >> 1;
        if (a[mid] < key) lo = mid + 1; else hi = mid;
    }
    return lo;
}

__global__ __launch_bounds__(128)
void segsum_tanh_kernel(const int* __restrict__ ngram_ids,
                        const int* __restrict__ segment_ids,
                        const float* __restrict__ ngram_w,
                        float* __restrict__ lang_f,
                        _Float16* __restrict__ lang_h,
                        int n_ngrams) {
    const int bag = blockIdx.x;
    const int t   = threadIdx.x;
    __shared__ int bounds[2];
    if (t < 2) bounds[t] = lower_bound_i32(segment_ids, n_ngrams, bag + t);
    __syncthreads();
    const int jbeg = bounds[0], jend = bounds[1];

    float acc = 0.0f;
    for (int j = jbeg; j < jend; ++j) {
        int gid = ngram_ids[j];
        acc += ngram_w[(size_t)gid * DIM + t];
    }
    float th = tanhf(acc);
    size_t idx = (size_t)bag * DIM + t;
    lang_f[idx] = th;
    lang_h[idx] = (_Float16)th;
}

// ---------------------------------------------------------------------------
// Kernel 2: latent_mat f32 -> f16 row-major [LATENT_P, DIM] + transposed
// [DIM, LATENT_P], both zero-padded.
// ---------------------------------------------------------------------------
__global__ __launch_bounds__(256)
void latent_cvt_kernel(const float* __restrict__ latent,
                       _Float16* __restrict__ latent_h,
                       _Float16* __restrict__ latentT_h) {
    int idx = blockIdx.x * 256 + threadIdx.x;
    int l = idx / DIM;
    int d = idx - l * DIM;
    float v = (l < LATENT) ? latent[(size_t)l * DIM + d] : 0.0f;
    latent_h[(size_t)l * DIM + d]       = (_Float16)v;
    latentT_h[(size_t)d * LATENT_P + l] = (_Float16)v;
}

// ---------------------------------------------------------------------------
// Kernel 3: fused  scores -> online softmax -> dist @ latent  (+ residual)
// 8 waves/block, each wave owns 16 rows.  B operands staged once per WG into
// double-buffered LDS via async global->LDS copies; copy of chunk c+1
// overlaps WMMA compute on chunk c.  Cross-lane work minimized: batched
// butterfly max-reduce, per-lane deferred row sums, rescale skipped unless a
// running max actually changed.
// ---------------------------------------------------------------------------
__device__ __forceinline__ v8f wmma_f16(v16h a, v16h b, v8f c) {
    return __builtin_amdgcn_wmma_f32_16x16x32_f16(
        false, a, false, b, (short)0, c, false, false);
}

__global__ __launch_bounds__(256)
void latent_softmax_kernel(const _Float16* __restrict__ lang_h,
                           const float* __restrict__ lang_f,
                           const _Float16* __restrict__ latent_h,
                           const _Float16* __restrict__ latentT_h,
                           float* __restrict__ token) {
    __shared__ _Float16 sBs [2][32 * BS_STRIDE];   // latent rows  (scores B)
    __shared__ _Float16 sBsT[2][128 * BST_STRIDE]; // latent^T     (P@V    B)
    __shared__ _Float16 sP  [8][16 * P_STRIDE];    // per-wave exp(P) tiles

    const int tid  = threadIdx.x;
    const int lane = tid & 31;
    const int wave = tid >> 5;
    const int m0   = (blockIdx.x * 8 + wave) * 16;
    _Float16* pA   = sP[wave];

    const int n    = lane & 15;
    const int koff = (lane >> 4) * 16;
    const int mrow = (lane >> 4) << 3;

    // staging indices (16-B segments, 2 per thread per buffer)
    const int r0a = tid >> 4,          c0a = tid & 15;
    const int r0b = (tid + 256) >> 4,  c0b = tid & 15;
    const int d1a = tid >> 2,          c1a = tid & 3;
    const int d1b = (tid + 256) >> 2,  c1b = tid & 3;

    // Persistent A tile: 16 rows x 128 K
    v16h a[4];
#pragma unroll
    for (int kb = 0; kb < 4; ++kb)
        a[kb] = *(const v16h*)(lang_h + (size_t)(m0 + n) * DIM + kb * 32 + koff);

    v8f acc[8];
    float mrun[8], lsum[8];
#pragma unroll
    for (int t = 0; t < 8; ++t) acc[t] = (v8f){0, 0, 0, 0, 0, 0, 0, 0};
#pragma unroll
    for (int r = 0; r < 8; ++r) { mrun[r] = -1.0e30f; lsum[r] = 0.0f; }

    // --- stage chunk 0 ---
    {
        const int l0 = 0;
        stage16(latent_h  + (size_t)(l0 + r0a) * DIM + c0a * 8, &sBs[0][r0a * BS_STRIDE + c0a * 8]);
        stage16(latent_h  + (size_t)(l0 + r0b) * DIM + c0b * 8, &sBs[0][r0b * BS_STRIDE + c0b * 8]);
        stage16(latentT_h + (size_t)d1a * LATENT_P + l0 + c1a * 8, &sBsT[0][d1a * BST_STRIDE + c1a * 8]);
        stage16(latentT_h + (size_t)d1b * LATENT_P + l0 + c1b * 8, &sBsT[0][d1b * BST_STRIDE + c1b * 8]);
    }
    wait_async_lds();
    __syncthreads();

    int buf = 0;
    for (int c = 0; c < NCHUNK; ++c) {
        const int l0 = c * 32;
        // --- stage chunk c+1 into the other buffer (overlaps compute) ---
        if (c + 1 < NCHUNK) {
            const int ln = l0 + 32;
            stage16(latent_h  + (size_t)(ln + r0a) * DIM + c0a * 8, &sBs[buf ^ 1][r0a * BS_STRIDE + c0a * 8]);
            stage16(latent_h  + (size_t)(ln + r0b) * DIM + c0b * 8, &sBs[buf ^ 1][r0b * BS_STRIDE + c0b * 8]);
            stage16(latentT_h + (size_t)d1a * LATENT_P + ln + c1a * 8, &sBsT[buf ^ 1][d1a * BST_STRIDE + c1a * 8]);
            stage16(latentT_h + (size_t)d1b * LATENT_P + ln + c1b * 8, &sBsT[buf ^ 1][d1b * BST_STRIDE + c1b * 8]);
        }

        const _Float16* Bs  = sBs[buf];
        const _Float16* BsT = sBsT[buf];

        // --- scores: two 16x16 tiles over K=128 (B from LDS) ---
        v8f c0 = (v8f){0, 0, 0, 0, 0, 0, 0, 0};
        v8f c1 = (v8f){0, 0, 0, 0, 0, 0, 0, 0};
#pragma unroll
        for (int kb = 0; kb < 4; ++kb) {
            v16h b0 = *(const v16h*)(Bs + (n)      * BS_STRIDE + kb * 32 + koff);
            v16h b1 = *(const v16h*)(Bs + (n + 16) * BS_STRIDE + kb * 32 + koff);
            c0 = wmma_f16(a[kb], b0, c0);
            c1 = wmma_f16(a[kb], b1, c1);
        }
        // mask padded columns
        if (l0 + n >= LATENT) {
#pragma unroll
            for (int r = 0; r < 8; ++r) c0[r] = -1.0e30f;
        }
        if (l0 + 16 + n >= LATENT) {
#pragma unroll
            for (int r = 0; r < 8; ++r) c1[r] = -1.0e30f;
        }

        // --- chunk max per row: batched butterfly (one wait per step) ---
        float cm[8];
#pragma unroll
        for (int r = 0; r < 8; ++r) cm[r] = fmaxf(c0[r], c1[r]);
#pragma unroll
        for (int s = 1; s <= 8; s <<= 1) {
            float o[8];
#pragma unroll
            for (int r = 0; r < 8; ++r) o[r] = __shfl_xor(cm[r], s, 32);
#pragma unroll
            for (int r = 0; r < 8; ++r) cm[r] = fmaxf(cm[r], o[r]);
        }

        // --- rescale only if some running max changed (uniform branch) ---
        int changed = 0;
#pragma unroll
        for (int r = 0; r < 8; ++r) changed |= (cm[r] > mrun[r]);
        if (__any(changed)) {
#pragma unroll
            for (int r = 0; r < 8; ++r) {
                float nm    = fmaxf(mrun[r], cm[r]);
                float scale = exp2f((mrun[r] - nm) * LOG2E);
                lsum[r] *= scale;
#pragma unroll
                for (int t = 0; t < 8; ++t) acc[t][r] *= scale;
                mrun[r] = nm;
            }
        }

        // --- exp, per-lane deferred sums, stage P tile to LDS ---
#pragma unroll
        for (int r = 0; r < 8; ++r) {
            float p0 = exp2f((c0[r] - mrun[r]) * LOG2E);
            float p1 = exp2f((c1[r] - mrun[r]) * LOG2E);
            lsum[r] += p0 + p1;
            int m = r + mrow;
            pA[m * P_STRIDE + n]      = (_Float16)p0;
            pA[m * P_STRIDE + n + 16] = (_Float16)p1;
        }

        // --- P @ latent[l0:l0+32, :] : A from LDS P tile, B from LDS ---
        v16h a2 = *(const v16h*)(pA + n * P_STRIDE + koff);
#pragma unroll
        for (int t = 0; t < 8; ++t) {
            v16h b2 = *(const v16h*)(BsT + (t * 16 + n) * BST_STRIDE + koff);
            acc[t] = wmma_f16(a2, b2, acc[t]);
        }

        wait_async_lds();
        __syncthreads();
        buf ^= 1;
    }

    // --- single deferred cross-lane sum reduction ---
#pragma unroll
    for (int s = 1; s <= 8; s <<= 1) {
        float o[8];
#pragma unroll
        for (int r = 0; r < 8; ++r) o[r] = __shfl_xor(lsum[r], s, 32);
#pragma unroll
        for (int r = 0; r < 8; ++r) lsum[r] += o[r];
    }
    float rinv[8];
#pragma unroll
    for (int r = 0; r < 8; ++r) rinv[r] = 1.0f / lsum[r];

    // --- epilogue: token = lang + acc * (1/rowsum) ---
#pragma unroll
    for (int t = 0; t < 8; ++t) {
#pragma unroll
        for (int r = 0; r < 8; ++r) {
            int m = r + mrow;
            size_t idx = (size_t)(m0 + m) * DIM + t * 16 + n;
            token[idx] = lang_f[idx] + acc[t][r] * rinv[r];
        }
    }
}

// ---------------------------------------------------------------------------
// Kernel 4: final embedding gather
// ---------------------------------------------------------------------------
__global__ __launch_bounds__(128)
void gather_kernel(const int* __restrict__ x,
                   const float* __restrict__ special,
                   const float* __restrict__ token,
                   float* __restrict__ out) {
    int tok = blockIdx.x;
    int t   = threadIdx.x;
    int id  = x[tok];
    float v = (id < NSPEC) ? special[(size_t)id * DIM + t]
                           : token[(size_t)(id - NSPEC) * DIM + t];
    out[(size_t)tok * DIM + t] = v;
}

// ---------------------------------------------------------------------------
extern "C" void kernel_launch(void* const* d_in, const int* in_sizes, int n_in,
                              void* d_out, int out_size, void* d_ws, size_t ws_size,
                              hipStream_t stream) {
    const int*   x           = (const int*)d_in[0];
    const int*   ngram_ids   = (const int*)d_in[1];
    const int*   segment_ids = (const int*)d_in[2];
    const float* ngram_w     = (const float*)d_in[3];
    const float* latent      = (const float*)d_in[4];
    const float* special     = (const float*)d_in[5];
    float*       out         = (float*)d_out;

    const int n_ngrams = in_sizes[1];
    const int n_tokens = in_sizes[0];

    char* ws = (char*)d_ws;
    float*    lang_f    = (float*)(ws);                       // 16,384,000 B
    _Float16* lang_h    = (_Float16*)(ws + 16384000);         //  8,192,000 B
    _Float16* latent_h  = (_Float16*)(ws + 24576000);         //  2,564,096 B
    _Float16* latentT_h = (_Float16*)(ws + 27140096);         //  2,564,096 B
    float*    token     = (float*)(ws + 29704192);            // 16,384,000 B

    segsum_tanh_kernel<<<V_WORDS, 128, 0, stream>>>(
        ngram_ids, segment_ids, ngram_w, lang_f, lang_h, n_ngrams);

    latent_cvt_kernel<<<(LATENT_P * DIM) / 256, 256, 0, stream>>>(
        latent, latent_h, latentT_h);

    latent_softmax_kernel<<<V_WORDS / 128, 256, 0, stream>>>(
        lang_h, lang_f, latent_h, latentT_h, token);

    gather_kernel<<<n_tokens, 128, 0, stream>>>(x, special, token, out);
}
